// RelativeMultiHeadAttention_17549236372022
// MI455X (gfx1250) — compile-verified, matrix operand order checked by reading, not run on hardware
//
#include <hip/hip_runtime.h>
#include <hip/hip_bf16.h>

// ---------------- problem constants ----------------
#define B_SZ   2
#define Q_LEN  1024
#define M_LEN  1024          // memory length
#define K_LEN  2048          // Q + M
#define D_MOD  1024
#define H_N    16
#define DH_N   64

typedef __bf16 bf16_t;
typedef __attribute__((ext_vector_type(16))) __bf16 v16bf;
typedef __attribute__((ext_vector_type(8)))  float  v8f;
typedef __attribute__((ext_vector_type(4)))  int    v4i;
typedef __attribute__((address_space(1))) v4i as1_v4i;
typedef __attribute__((address_space(3))) v4i as3_v4i;

// ---------------- bf16 helpers ----------------
__device__ __forceinline__ bf16_t f2bf(float f) {
    unsigned u = __builtin_bit_cast(unsigned, f);
    unsigned r = u + 0x7FFFu + ((u >> 16) & 1u);      // round-to-nearest-even
    unsigned short h = (unsigned short)(r >> 16);
    return __builtin_bit_cast(bf16_t, h);
}
__device__ __forceinline__ float bf2f(bf16_t b) {
    unsigned short h = __builtin_bit_cast(unsigned short, b);
    unsigned u = ((unsigned)h) << 16;
    return __builtin_bit_cast(float, u);
}

// K-index map for 16-bit WMMA fragments (ISA 7.12.2, 16-bit A 16x32):
// lanes 0-15 : e 0..7 -> K 0..7,   e 8..15 -> K 16..23
// lanes 16-31: e 0..7 -> K 8..15,  e 8..15 -> K 24..31
__device__ __forceinline__ int kmap(int e, int half) {
    int k = (e < 8) ? e : (e + 8);
    return k + half * 8;
}

__device__ __forceinline__ v8f wmma_bf16(v16bf a, v16bf b, v8f c) {
    return __builtin_amdgcn_wmma_f32_16x16x32_bf16(
        /*neg_a=*/false, a, /*neg_b=*/false, b,
        /*c_mod=*/(short)0, c, /*reuse_a=*/false, /*reuse_b=*/false);
}

// ---------------- async global->LDS copy (CDNA5 GLOBAL_LOAD_ASYNC_TO_LDS_B128) ----------------
#if __has_builtin(__builtin_amdgcn_global_load_async_to_lds_b128)
#define HAVE_ASYNC_COPY 1
#else
#define HAVE_ASYNC_COPY 0
#endif

__device__ __forceinline__ void cp16(void* lds_dst, const void* gsrc) {
#if HAVE_ASYNC_COPY
    __builtin_amdgcn_global_load_async_to_lds_b128(
        (as1_v4i*)(uintptr_t)gsrc,
        (as3_v4i*)(unsigned)(uintptr_t)lds_dst,
        0, 0);
#else
    *(uint4*)lds_dst = *(const uint4*)gsrc;
#endif
}

__device__ __forceinline__ void cp_wait() {
#if HAVE_ASYNC_COPY
#if __has_builtin(__builtin_amdgcn_s_wait_asynccnt)
    __builtin_amdgcn_s_wait_asynccnt(0);
#else
    asm volatile("s_wait_asynccnt 0x0" ::: "memory");
#endif
#endif
}

// ---------------- fp32 -> bf16 convert (activations) ----------------
__global__ __launch_bounds__(256) void cvt_f32_bf16(const float* __restrict__ src,
                                                    bf16_t* __restrict__ dst, int n) {
    int i = (blockIdx.x * 256 + threadIdx.x) * 4;
    if (i + 3 < n) {
        float4 f = *(const float4*)(src + i);
        dst[i + 0] = f2bf(f.x);
        dst[i + 1] = f2bf(f.y);
        dst[i + 2] = f2bf(f.z);
        dst[i + 3] = f2bf(f.w);
    }
}

// ---------------- fp32 -> bf16 convert + transpose (weights) ----------------
// W[Ka][N] fp32 row-major  ->  Wt[N][Ka] bf16 row-major
__global__ __launch_bounds__(256) void cvt_transpose_bf16(const float* __restrict__ W,
                                                          bf16_t* __restrict__ Wt,
                                                          int Ka, int N) {
    __shared__ bf16_t tile[32][33];
    const int n0 = blockIdx.x * 32, k0 = blockIdx.y * 32;
    const int tx = threadIdx.x & 31, ty = threadIdx.x >> 5;   // 32 x 8
#pragma unroll
    for (int s = 0; s < 4; ++s) {
        int k = ty + s * 8;
        tile[k][tx] = f2bf(W[(size_t)(k0 + k) * N + n0 + tx]);
    }
    __syncthreads();
#pragma unroll
    for (int s = 0; s < 4; ++s) {
        int n = ty + s * 8;
        Wt[(size_t)(n0 + n) * Ka + k0 + tx] = tile[tx][n];
    }
}

// ---------------- bf16 transpose of the V head-matrix ----------------
// w[B*K][3D] (v block at col D) -> Vt[b][d][k]   ([B, D, K])
__global__ __launch_bounds__(256) void transpose_v_bf16(const bf16_t* __restrict__ w,
                                                        bf16_t* __restrict__ vt) {
    __shared__ bf16_t tile[32][33];
    const int b  = blockIdx.z;
    const int d0 = blockIdx.x * 32, k0 = blockIdx.y * 32;
    const int tx = threadIdx.x & 31, ty = threadIdx.x >> 5;
#pragma unroll
    for (int s = 0; s < 4; ++s) {
        int k = ty + s * 8;
        tile[k][tx] = w[((size_t)b * K_LEN + k0 + k) * (3 * D_MOD) + D_MOD + d0 + tx];
    }
    __syncthreads();
#pragma unroll
    for (int s = 0; s < 4; ++s) {
        int d = ty + s * 8;
        vt[((size_t)b * D_MOD + d0 + d) * K_LEN + k0 + tx] = tile[tx][d];
    }
}

// ---------------- generic WMMA GEMM: C[M,N] = A[M,Ka] * Wt[N,Ka]^T + bias ----------------
#define GM_BM 128
#define GM_BN 64
#define GM_BK 64

__global__ __launch_bounds__(256) void gemm_bf16_wmma(
    const bf16_t* __restrict__ A, const bf16_t* __restrict__ Bt,
    const float* __restrict__ bias,
    bf16_t* __restrict__ out_bf, float* __restrict__ out_f,
    int Ka, int N) {
    __shared__ __align__(16) bf16_t As[GM_BM][GM_BK];   // 16 KB, m-major
    __shared__ __align__(16) bf16_t Bs[GM_BN][GM_BK];   //  8 KB, n-major
    const int bm0  = blockIdx.y * GM_BM;
    const int bn0  = blockIdx.x * GM_BN;
    const int tid  = threadIdx.x;
    const int wave = tid >> 5;
    const int lane = tid & 31;
    const int half = lane >> 4;
    const int lr   = lane & 15;

    v8f acc[4] = {};

    for (int kc = 0; kc < Ka; kc += GM_BK) {
        // A tile: 128x64 bf16 = 1024 16B-chunks, 4 async copies per thread
#pragma unroll
        for (int t = 0; t < 4; ++t) {
            int c = tid + t * 256;
            int row = c >> 3, col = (c & 7) * 8;
            cp16(&As[row][col], &A[(size_t)(bm0 + row) * Ka + kc + col]);
        }
        // B tile: 64x64 bf16 = 512 chunks, 2 per thread
#pragma unroll
        for (int t = 0; t < 2; ++t) {
            int c = tid + t * 256;
            int row = c >> 3, col = (c & 7) * 8;
            cp16(&Bs[row][col], &Bt[(size_t)(bn0 + row) * Ka + kc + col]);
        }
        if (kc + GM_BK < Ka) {   // emits global_prefetch_b8
            int row = tid >> 3, col = (tid & 7) * 8;
            __builtin_prefetch(&A[(size_t)(bm0 + row) * Ka + kc + GM_BK + col], 0, 3);
            __builtin_prefetch(&Bt[(size_t)(bn0 + row) * Ka + kc + GM_BK + col], 0, 3);
        }
        cp_wait();
        __syncthreads();

        v16bf af[2];
#pragma unroll
        for (int kk = 0; kk < 2; ++kk)
#pragma unroll
            for (int e = 0; e < 16; ++e)
                af[kk][e] = As[wave * 16 + lr][kk * 32 + kmap(e, half)];
#pragma unroll
        for (int n = 0; n < 4; ++n) {
#pragma unroll
            for (int kk = 0; kk < 2; ++kk) {
                v16bf bfv;
#pragma unroll
                for (int e = 0; e < 16; ++e)
                    bfv[e] = Bs[n * 16 + lr][kk * 32 + kmap(e, half)];
                acc[n] = wmma_bf16(af[kk], bfv, acc[n]);
            }
        }
        __syncthreads();
    }

#pragma unroll
    for (int n = 0; n < 4; ++n) {
        int col = bn0 + n * 16 + lr;
        float bv = bias ? bias[col] : 0.f;
#pragma unroll
        for (int r = 0; r < 8; ++r) {
            int row = bm0 + wave * 16 + r + 8 * half;
            float vv = acc[n][r] + bv;
            if (out_f) out_f[(size_t)row * N + col] = vv;
            else       out_bf[(size_t)row * N + col] = f2bf(vv);
        }
    }
}

// ---------------- flash-style rel-attention ----------------
// 4 waves / block; block owns 64 q-rows, waves share K / V^T / R tiles in LDS.
// score(i,j) = [ (q_i+u)·k_j  +  (q_i+v)·r_{j+Q-1-i} ] / 8,  mask j <= M+i.
__global__ __launch_bounds__(128) void relattn_wmma(
    const bf16_t* __restrict__ w,    // [B*K, 3D]  (q | v | k column blocks)
    const bf16_t* __restrict__ rp,   // [B*K, D]
    const bf16_t* __restrict__ vt,   // [B, D, K]
    const float*  __restrict__ u_,   // [H*DH]
    const float*  __restrict__ v_,   // [H*DH]
    bf16_t* __restrict__ attn) {     // [B*Q, D]
    const int i0blk = blockIdx.x * 64;
    const int h     = blockIdx.y;
    const int b     = blockIdx.z;
    const int tid   = threadIdx.x;
    const int wv    = tid >> 5;          // wave id 0..3
    const int lane  = tid & 31;
    const int half  = lane >> 4;
    const int lr    = lane & 15;
    const int i0    = i0blk + wv * 16;   // this wave's q tile

    __shared__ __align__(16) bf16_t Qs[64][64];    //  8 KB (block q rows)
    __shared__ __align__(16) bf16_t Ks[32][64];    //  4 KB [key][dh]
    __shared__ __align__(16) bf16_t Vts[64][32];   //  4 KB [dh][key]
    __shared__ __align__(16) bf16_t Rs[96][64];    // 12 KB [m][dh], union of 4 shift bands
    __shared__ float  RAW[4][16][48];              // 12 KB per-wave BD scratch
    __shared__ __align__(16) bf16_t Ps[4][16][32]; //  4 KB per-wave P tiles

    const size_t rowbase = (size_t)b * K_LEN;
    const int    hcol    = h * DH_N;
    const int    W3      = 3 * D_MOD;

    // q rows i0blk..i0blk+63  (512 chunks, 4 per thread)
    for (int c = tid; c < 64 * 8; c += 128) {
        int row = c >> 3, col = (c & 7) * 8;
        cp16(&Qs[row][col], &w[(rowbase + M_LEN + i0blk + row) * W3 + hcol + col]);
    }
    cp_wait();
    __syncthreads();

    // A fragments: (q+u) and (q+v), two dh-chunks of 32
    v16bf aqu[2], aqv[2];
#pragma unroll
    for (int kc2 = 0; kc2 < 2; ++kc2) {
#pragma unroll
        for (int e = 0; e < 16; ++e) {
            int d = kc2 * 32 + kmap(e, half);
            float qf = bf2f(Qs[wv * 16 + lr][d]);
            aqu[kc2][e] = f2bf(qf + u_[hcol + d]);
            aqv[kc2][e] = f2bf(qf + v_[hcol + d]);
        }
    }

    v8f acc[4] = {};
    float mrun[8], lrun[8];
#pragma unroll
    for (int r = 0; r < 8; ++r) { mrun[r] = -1e30f; lrun[r] = 0.f; }

    const int nkeys = M_LEN + i0blk + 64;     // uniform across waves
    const int rbase = 48 - 16 * wv;           // this wave's window into Rs

    for (int j0 = 0; j0 < nkeys; j0 += 32) {
        __syncthreads();   // previous iter LDS reads complete before overwrite
        // ---- stage K (32x64), V^T (64x32), R band (96x64) ----
        for (int c = tid; c < 32 * 8; c += 128) {
            int row = c >> 3, col = (c & 7) * 8;
            cp16(&Ks[row][col],
                 &w[(rowbase + j0 + row) * W3 + 2 * D_MOD + hcol + col]);
        }
        for (int c = tid; c < 64 * 4; c += 128) {
            int row = c >> 2, col = (c & 3) * 8;
            cp16(&Vts[row][col],
                 &vt[((size_t)b * D_MOD + hcol + row) * K_LEN + j0 + col]);
        }
        const int m0 = j0 + Q_LEN - 64 - i0blk;   // lowest band row (wave 3), >= 0
        for (int c = tid; c < 96 * 8; c += 128) {
            int row = c >> 3, col = (c & 7) * 8;
            int mr = m0 + row; if (mr >= K_LEN) mr = K_LEN - 1;  // masked region only
            cp16(&Rs[row][col], &rp[(rowbase + mr) * D_MOD + hcol + col]);
        }
        cp_wait();
        __syncthreads();

        // ---- AC: (q+u) x K^T ----
        v8f s0 = {}, s1 = {};
#pragma unroll
        for (int kc2 = 0; kc2 < 2; ++kc2) {
            v16bf b0, b1;
#pragma unroll
            for (int e = 0; e < 16; ++e) {
                int dh = kc2 * 32 + kmap(e, half);
                b0[e] = Ks[lr][dh];
                b1[e] = Ks[16 + lr][dh];
            }
            s0 = wmma_bf16(aqu[kc2], b0, s0);
            s1 = wmma_bf16(aqu[kc2], b1, s1);
        }

        // ---- BD raw: (q+v) x R^T over this wave's 48-wide m band ----
        v8f raw[3] = {};
#pragma unroll
        for (int mt = 0; mt < 3; ++mt) {
#pragma unroll
            for (int kc2 = 0; kc2 < 2; ++kc2) {
                v16bf br;
#pragma unroll
                for (int e = 0; e < 16; ++e)
                    br[e] = Rs[rbase + mt * 16 + lr][kc2 * 32 + kmap(e, half)];
                raw[mt] = wmma_bf16(aqv[kc2], br, raw[mt]);
            }
        }
#pragma unroll
        for (int mt = 0; mt < 3; ++mt)
#pragma unroll
            for (int r = 0; r < 8; ++r)
                RAW[wv][r + 8 * half][mt * 16 + lr] = raw[mt][r];
        __syncthreads();

        // ---- rel-shift gather + scale + causal mask ----
        float sc0[8], sc1[8];
#pragma unroll
        for (int r = 0; r < 8; ++r) {
            int irow = r + 8 * half;
            int iabs = i0 + irow;
            int shift = 15 - irow;
            float v0 = (s0[r] + RAW[wv][irow][lr + shift]) * 0.125f;
            float v1 = (s1[r] + RAW[wv][irow][16 + lr + shift]) * 0.125f;
            if (j0 + lr      > M_LEN + iabs) v0 = -1e30f;
            if (j0 + 16 + lr > M_LEN + iabs) v1 = -1e30f;
            sc0[r] = v0; sc1[r] = v1;
        }

        // ---- online softmax (reduce across the 16 lanes of each half) ----
#pragma unroll
        for (int r = 0; r < 8; ++r) {
            float tmax = fmaxf(sc0[r], sc1[r]);
#pragma unroll
            for (int mk = 1; mk < 16; mk <<= 1)
                tmax = fmaxf(tmax, __shfl_xor(tmax, mk, 32));
            float mn    = fmaxf(mrun[r], tmax);
            float alpha = __expf(mrun[r] - mn);
            mrun[r] = mn;
            float p0 = __expf(sc0[r] - mn);
            float p1 = __expf(sc1[r] - mn);
            Ps[wv][r + 8 * half][lr]      = f2bf(p0);
            Ps[wv][r + 8 * half][16 + lr] = f2bf(p1);
            float rs = p0 + p1;
#pragma unroll
            for (int mk = 1; mk < 16; mk <<= 1)
                rs += __shfl_xor(rs, mk, 32);
            lrun[r] = lrun[r] * alpha + rs;
#pragma unroll
            for (int n = 0; n < 4; ++n) acc[n][r] *= alpha;
        }
        __syncthreads();

        // ---- PV: P[16x32] x V[32x64]  (B-frag = Vt rows, contiguous) ----
        v16bf ap;
#pragma unroll
        for (int e = 0; e < 16; ++e) ap[e] = Ps[wv][lr][kmap(e, half)];
#pragma unroll
        for (int n = 0; n < 4; ++n) {
            v16bf bv;
#pragma unroll
            for (int e = 0; e < 16; ++e) bv[e] = Vts[n * 16 + lr][kmap(e, half)];
            acc[n] = wmma_bf16(ap, bv, acc[n]);
        }
    }

    // ---- normalize + write bf16 [B*Q, D] ----
#pragma unroll
    for (int n = 0; n < 4; ++n) {
#pragma unroll
        for (int r = 0; r < 8; ++r) {
            int row = i0 + r + 8 * half;
            attn[((size_t)b * Q_LEN + row) * D_MOD + hcol + n * 16 + lr] =
                f2bf(acc[n][r] / lrun[r]);
        }
    }
}

// ---------------- workspace layout (bf16 elements) ----------------
constexpr size_t OFF_CAT  = 0;                                           // [B*K, D]
constexpr size_t OFF_RB   = OFF_CAT  + (size_t)B_SZ * K_LEN * D_MOD;     // [B*K, D]
constexpr size_t OFF_WQKV = OFF_RB   + (size_t)B_SZ * K_LEN * D_MOD;     // [3D, D]  (transposed)
constexpr size_t OFF_WR   = OFF_WQKV + (size_t)D_MOD * 3 * D_MOD;        // [D, D]   (transposed)
constexpr size_t OFF_WO   = OFF_WR   + (size_t)D_MOD * D_MOD;            // [D, D]   (transposed)
constexpr size_t OFF_W    = OFF_WO   + (size_t)D_MOD * D_MOD;            // [B*K, 3D]
constexpr size_t OFF_RP   = OFF_W    + (size_t)B_SZ * K_LEN * 3 * D_MOD; // [B*K, D]
constexpr size_t OFF_ATT  = OFF_RP   + (size_t)B_SZ * K_LEN * D_MOD;     // [B*Q, D]
constexpr size_t OFF_VT   = OFF_ATT  + (size_t)B_SZ * Q_LEN * D_MOD;     // [B, D, K]

extern "C" void kernel_launch(void* const* d_in, const int* in_sizes, int n_in,
                              void* d_out, int out_size, void* d_ws, size_t ws_size,
                              hipStream_t stream) {
    (void)in_sizes; (void)n_in; (void)out_size; (void)ws_size;
    const float* inputs     = (const float*)d_in[0];
    const float* inputs_mem = (const float*)d_in[1];
    const float* r_in       = (const float*)d_in[2];
    const float* W_qkv      = (const float*)d_in[3];
    const float* b_qkv      = (const float*)d_in[4];
    const float* W_r        = (const float*)d_in[5];
    const float* b_r        = (const float*)d_in[6];
    const float* W_o        = (const float*)d_in[7];
    const float* b_o        = (const float*)d_in[8];
    const float* u_p        = (const float*)d_in[9];
    const float* v_p        = (const float*)d_in[10];

    bf16_t* ws = (bf16_t*)d_ws;
    auto cvt = [&](const float* src, bf16_t* dst, size_t n) {
        cvt_f32_bf16<<<dim3((unsigned)(n / 1024)), dim3(256), 0, stream>>>(src, dst, (int)n);
    };

    // 1) convert activations; convert + transpose weights
    for (int b = 0; b < B_SZ; ++b) {
        cvt(inputs_mem + (size_t)b * M_LEN * D_MOD,
            ws + OFF_CAT + (size_t)b * K_LEN * D_MOD, (size_t)M_LEN * D_MOD);
        cvt(inputs + (size_t)b * Q_LEN * D_MOD,
            ws + OFF_CAT + (size_t)b * K_LEN * D_MOD + (size_t)M_LEN * D_MOD,
            (size_t)Q_LEN * D_MOD);
    }
    cvt(r_in, ws + OFF_RB, (size_t)B_SZ * K_LEN * D_MOD);
    cvt_transpose_bf16<<<dim3(3 * D_MOD / 32, D_MOD / 32), dim3(256), 0, stream>>>(
        W_qkv, ws + OFF_WQKV, D_MOD, 3 * D_MOD);
    cvt_transpose_bf16<<<dim3(D_MOD / 32, D_MOD / 32), dim3(256), 0, stream>>>(
        W_r, ws + OFF_WR, D_MOD, D_MOD);
    cvt_transpose_bf16<<<dim3(D_MOD / 32, D_MOD / 32), dim3(256), 0, stream>>>(
        W_o, ws + OFF_WO, D_MOD, D_MOD);

    // 2) w = cat @ W_qkv + b_qkv   [4096 x 3072]
    gemm_bf16_wmma<<<dim3(3 * D_MOD / GM_BN, B_SZ * K_LEN / GM_BM), dim3(256), 0, stream>>>(
        ws + OFF_CAT, ws + OFF_WQKV, b_qkv, ws + OFF_W, nullptr, D_MOD, 3 * D_MOD);

    // 3) rp = r @ W_r + b_r        [4096 x 1024]
    gemm_bf16_wmma<<<dim3(D_MOD / GM_BN, B_SZ * K_LEN / GM_BM), dim3(256), 0, stream>>>(
        ws + OFF_RB, ws + OFF_WR, b_r, ws + OFF_RP, nullptr, D_MOD, D_MOD);

    // 3b) Vt[b][d][k] = w_v[b][k][d]  (makes PV B-fragments row-contiguous)
    transpose_v_bf16<<<dim3(D_MOD / 32, K_LEN / 32, B_SZ), dim3(256), 0, stream>>>(
        ws + OFF_W, ws + OFF_VT);

    // 4) attention: 4 waves / block, 64 q-rows per block
    relattn_wmma<<<dim3(Q_LEN / 64, H_N, B_SZ), dim3(128), 0, stream>>>(
        ws + OFF_W, ws + OFF_RP, ws + OFF_VT, u_p, v_p, ws + OFF_ATT);

    // 5) out = attn @ W_o + b_o -> fp32 d_out   [2048 x 1024]
    gemm_bf16_wmma<<<dim3(D_MOD / GM_BN, B_SZ * Q_LEN / GM_BM), dim3(256), 0, stream>>>(
        ws + OFF_ATT, ws + OFF_WO, b_o, nullptr, (float*)d_out, D_MOD, D_MOD);
}
